// TransformerBlockQuantum_65481071402579
// MI455X (gfx1250) — compile-verified, matrix operand order checked by reading, not run on hardware
//
#include <hip/hip_runtime.h>
#include <hip/hip_bf16.h>

// MI455X (gfx1250) fused transformer block.
//  - Compute-bound GEMMs on v_wmma_f32_16x16x32_bf16 (f32 accumulate).
//  - Tile staging via the Tensor Data Mover (tensor_load_to_lds, TENSORcnt),
//    double-buffered, with D# LDS padding giving a conflict-free 80B row stride.
//  - Attention scores+softmax fused per (b,h); softmax rows reduced with
//    16-lane shfl_xor (each 128-wide row lives in one 16-lane half per C layout).
//  - v / wv / bv are dead in the reference and skipped.

typedef __attribute__((ext_vector_type(16))) __bf16 v16bf;
typedef __attribute__((ext_vector_type(8)))  float  v8f;
typedef __attribute__((ext_vector_type(4)))  unsigned int v4u;
typedef __attribute__((ext_vector_type(8)))  int    v8i;
typedef __attribute__((ext_vector_type(4)))  int    v4i;

struct alignas(16) U128 { unsigned int w[4]; };
union FragAB {
  U128  u[2];
  v16bf v;
};

#if defined(__has_builtin)
#  if __has_builtin(__builtin_amdgcn_tensor_load_to_lds) && \
      __has_builtin(__builtin_amdgcn_s_wait_tensorcnt)
#    define USE_TDM 1
#  endif
#endif
#ifndef USE_TDM
#  define USE_TDM 0
#endif

__device__ __forceinline__ unsigned short f32_to_bf16_rne(float f) {
  unsigned int u = __float_as_uint(f);
  u += 0x7FFFu + ((u >> 16) & 1u);   // round-to-nearest-even
  return (unsigned short)(u >> 16);
}

#if USE_TDM
// Per CDNA5 aperture rules, the LDS byte address is the low 32 bits of the
// generic (flat) address of a __shared__ object.
__device__ __forceinline__ unsigned lds_addr_of(const void* p) {
  return (unsigned)(unsigned long long)(size_t)p;
}

// Issue a 2-D TENSOR_LOAD_TO_LDS: tile_h rows of tile_w bf16 elements, global
// row stride `row_stride` elements, optional LDS padding (interval/amount codes
// per D# group1: interval code c -> 2^(c+1) DWORDs, amount code a -> (a+1) DWORDs).
__device__ __forceinline__ void tdm_load_2d_bf16(const void* gaddr, unsigned lds_addr,
                                                 int tile_w, int tile_h,
                                                 long long row_stride,
                                                 int pad_interval_code,
                                                 int pad_amount_code,
                                                 int pad_enable) {
  const unsigned long long ga = (unsigned long long)(size_t)gaddr;
  v4u g0;
  g0[0] = 1u;                                            // count=1 (valid user D#)
  g0[1] = lds_addr;                                      // LDS byte address
  g0[2] = (unsigned)(ga & 0xFFFFFFFFu);                  // global_addr[31:0]
  g0[3] = (unsigned)((ga >> 32) & 0x1FFFFFFu) | (2u << 30);  // global_addr[56:32], type=2

  const unsigned td0 = 0x7FFFFFFFu;                      // tensor_dim0 (no OOB clip)
  const unsigned td1 = 0x7FFFFFFFu;                      // tensor_dim1
  const unsigned long long s0 = (unsigned long long)row_stride;
  v8i g1;
  g1[0] = (int)((1u << 16) |                             // data_size = 2 bytes (bf16)
                ((unsigned)pad_enable << 20) |
                ((unsigned)pad_interval_code << 22) |
                ((unsigned)pad_amount_code << 25));
  g1[1] = (int)((td0 & 0xFFFFu) << 16);                  // tensor_dim0[15:0]
  g1[2] = (int)(((td0 >> 16) & 0xFFFFu) | ((td1 & 0xFFFFu) << 16));
  g1[3] = (int)(((td1 >> 16) & 0xFFFFu) | ((unsigned)tile_w << 16));  // tile_dim0
  g1[4] = (int)((unsigned)tile_h);                       // tile_dim1 (tile_dim2 = 0)
  g1[5] = (int)(unsigned)(s0 & 0xFFFFFFFFu);             // tensor_dim0_stride[31:0]
  g1[6] = (int)(unsigned)((s0 >> 32) & 0xFFFFu);         // stride[47:32], dim1_stride=0
  g1[7] = 0;
  const v4i z4 = (v4i){0, 0, 0, 0};                      // 2-D tensor: groups 2/3 unused
#if __clang_major__ >= 23
  const v8i z8 = (v8i){0, 0, 0, 0, 0, 0, 0, 0};
  __builtin_amdgcn_tensor_load_to_lds(g0, g1, z4, z4, z8, 0);
#else
  __builtin_amdgcn_tensor_load_to_lds(g0, g1, z4, z4, 0);
#endif
}
#endif  // USE_TDM

// ---------------------------------------------------------------- cvt f32->bf16
__global__ __launch_bounds__(256)
void cvt_f32_to_bf16_kernel(const float* __restrict__ in,
                            unsigned short* __restrict__ out, size_t n4) {
  size_t i = (size_t)blockIdx.x * 256 + threadIdx.x;
  if (i >= n4) return;
  const float4 v = reinterpret_cast<const float4*>(in)[i];
  uint2 p;
  p.x = (unsigned int)f32_to_bf16_rne(v.x) | ((unsigned int)f32_to_bf16_rne(v.y) << 16);
  p.y = (unsigned int)f32_to_bf16_rne(v.z) | ((unsigned int)f32_to_bf16_rne(v.w) << 16);
  reinterpret_cast<uint2*>(out)[i] = p;
}

// ------------------------------------------------- transpose + cvt: WT[n][k] = W[k][n]
__global__ __launch_bounds__(256)
void transpose_cvt_kernel(const float* __restrict__ W,
                          unsigned short* __restrict__ WT, int K, int N) {
  __shared__ float tile[32][33];
  const int n0 = blockIdx.x * 32, k0 = blockIdx.y * 32;
  const int tx = threadIdx.x, ty = threadIdx.y;   // block (32,8)
#pragma unroll
  for (int i = 0; i < 32; i += 8)
    tile[ty + i][tx] = W[(size_t)(k0 + ty + i) * N + (n0 + tx)];
  __syncthreads();
#pragma unroll
  for (int i = 0; i < 32; i += 8)
    WT[(size_t)(n0 + ty + i) * K + (k0 + tx)] = f32_to_bf16_rne(tile[tx][ty + i]);
}

// ---------------------------------------------------------------- WMMA GEMM
// C[M,N] = A[M,K](bf16) * B (given as BT[N,K] bf16) + bias, optional relu.
// 128x128 tile per 256-thread block (8 waves, 32x64 per wave, 2x4 WMMA tiles).
// K slabs of 32, double-buffered in LDS, staged by the TDM.
#define GBM 128
#define GBN 128
#define GBK 32
#define LDST 40   // LDS row stride in ushorts: 32 data + 4-DWORD TDM pad = 80 B

__global__ __launch_bounds__(256)
void wmma_gemm_bf16_kernel(const unsigned short* __restrict__ A,
                           const unsigned short* __restrict__ BT,
                           const float* __restrict__ bias,
                           float* __restrict__ Cf,
                           unsigned short* __restrict__ Cb,
                           int M, int N, int K, int relu) {
  __shared__ alignas(16) unsigned short a_lds[2][GBM][LDST];
  __shared__ alignas(16) unsigned short b_lds[2][GBN][LDST];

  const int tid   = threadIdx.x;
  const int lane  = tid & 31;
  const int wave  = tid >> 5;
  const int l16   = lane & 15;
  const int half  = lane >> 4;
  const int waveM = wave >> 1;    // 0..3 -> 32-row strip
  const int waveN = wave & 1;     // 0..1 -> 64-col strip

  const int tileM = blockIdx.y * GBM;
  const int tileN = blockIdx.x * GBN;

  v8f acc[2][4];
#pragma unroll
  for (int i = 0; i < 2; ++i)
#pragma unroll
    for (int j = 0; j < 4; ++j)
      acc[i][j] = v8f{0.f, 0.f, 0.f, 0.f, 0.f, 0.f, 0.f, 0.f};

  const int kTiles = K / GBK;

#if USE_TDM
  // Prologue: DMA slab 0 into buffer 0 (wave 0 only; TDM ignores EXEC, one issue/wave).
  if (tid < 32) {
    tdm_load_2d_bf16(A  + (size_t)tileM * K, lds_addr_of(&a_lds[0][0][0]),
                     GBK, GBM, K, /*interval 16 DW*/3, /*pad 4 DW*/3, 1);
    tdm_load_2d_bf16(BT + (size_t)tileN * K, lds_addr_of(&b_lds[0][0][0]),
                     GBK, GBN, K, 3, 3, 1);
  }
#endif

  for (int kt = 0; kt < kTiles; ++kt) {
    const int cur = kt & 1;
#if USE_TDM
    __syncthreads();   // buffer cur^1 is free (readers of slab kt-1 done)
    if (tid < 32) {
      if (kt + 1 < kTiles) {
        tdm_load_2d_bf16(A  + (size_t)tileM * K + (size_t)(kt + 1) * GBK,
                         lds_addr_of(&a_lds[cur ^ 1][0][0]), GBK, GBM, K, 3, 3, 1);
        tdm_load_2d_bf16(BT + (size_t)tileN * K + (size_t)(kt + 1) * GBK,
                         lds_addr_of(&b_lds[cur ^ 1][0][0]), GBK, GBN, K, 3, 3, 1);
        __builtin_amdgcn_s_wait_tensorcnt(2);   // slab kt landed; kt+1 in flight
      } else {
        __builtin_amdgcn_s_wait_tensorcnt(0);
      }
    }
    __syncthreads();   // buffer cur ready for all waves
#else
    // Fallback: software staging; issue all global loads before any LDS store.
    __syncthreads();
    {
      const int c0 = tid, c1 = tid + 256;        // 512 16B chunks per matrix
      const int r0 = c0 >> 2, p0 = c0 & 3;
      const int r1 = c1 >> 2, p1 = c1 & 3;
      U128 a0 = *reinterpret_cast<const U128*>(A + (size_t)(tileM + r0) * K + (size_t)kt * GBK + p0 * 8);
      U128 a1 = *reinterpret_cast<const U128*>(A + (size_t)(tileM + r1) * K + (size_t)kt * GBK + p1 * 8);
      U128 b0 = *reinterpret_cast<const U128*>(BT + (size_t)(tileN + r0) * K + (size_t)kt * GBK + p0 * 8);
      U128 b1 = *reinterpret_cast<const U128*>(BT + (size_t)(tileN + r1) * K + (size_t)kt * GBK + p1 * 8);
      *reinterpret_cast<U128*>(&a_lds[cur][r0][p0 * 8]) = a0;
      *reinterpret_cast<U128*>(&a_lds[cur][r1][p1 * 8]) = a1;
      *reinterpret_cast<U128*>(&b_lds[cur][r0][p0 * 8]) = b0;
      *reinterpret_cast<U128*>(&b_lds[cur][r1][p1 * 8]) = b1;
    }
    __syncthreads();
#endif

    FragAB af[2], bf[4];
#pragma unroll
    for (int ms = 0; ms < 2; ++ms) {  // ISA 16-bit A layout: lane=M%16, half picks K-octet
      const int m = waveM * 32 + ms * 16 + l16;
      const unsigned short* p = &a_lds[cur][m][half * 8];
      af[ms].u[0] = *reinterpret_cast<const U128*>(p);        // K 0..7   (+half*8)
      af[ms].u[1] = *reinterpret_cast<const U128*>(p + 16);   // K 16..23 (+half*8)
    }
#pragma unroll
    for (int ns = 0; ns < 4; ++ns) {  // BT rows are N -> same fragment shape
      const int n = waveN * 64 + ns * 16 + l16;
      const unsigned short* p = &b_lds[cur][n][half * 8];
      bf[ns].u[0] = *reinterpret_cast<const U128*>(p);
      bf[ns].u[1] = *reinterpret_cast<const U128*>(p + 16);
    }
#pragma unroll
    for (int ms = 0; ms < 2; ++ms)
#pragma unroll
      for (int ns = 0; ns < 4; ++ns)
        acc[ms][ns] = __builtin_amdgcn_wmma_f32_16x16x32_bf16(
            false, af[ms].v, false, bf[ns].v, (short)0, acc[ms][ns], false, false);
  }

  // Epilogue: C layout -> m = r + 8*half, n = lane&15 within each 16x16 tile.
#pragma unroll
  for (int ms = 0; ms < 2; ++ms) {
#pragma unroll
    for (int ns = 0; ns < 4; ++ns) {
      const int nG = tileN + waveN * 64 + ns * 16 + l16;
      const float bb = bias ? bias[nG] : 0.f;
#pragma unroll
      for (int r = 0; r < 8; ++r) {
        const int mG = tileM + waveM * 32 + ms * 16 + half * 8 + r;
        float v = acc[ms][ns][r] + bb;
        if (relu) v = fmaxf(v, 0.f);
        const size_t idx = (size_t)mG * N + nG;
        if (Cf) Cf[idx] = v;
        if (Cb) Cb[idx] = f32_to_bf16_rne(v);
      }
    }
  }
}

// ---------------------------------------------- fused scores + softmax per (b,h)
// One 128-thread block (4 waves) per head: P[b*T+tq][h*T+tk] = softmax(q.k^T/sqrt(DK)).
__global__ __launch_bounds__(128)
void attn_scores_softmax_kernel(const unsigned short* __restrict__ Q,
                                const unsigned short* __restrict__ Km,
                                unsigned short* __restrict__ P,
                                int T_, int E_, int H_) {
  __shared__ alignas(16) unsigned short q_lds[128][128];
  __shared__ alignas(16) unsigned short k_lds[128][128];
  const int bh = blockIdx.x;
  const int b  = bh / H_;
  const int h  = bh % H_;
  const int DK = E_ / H_;            // 128
  const int tid  = threadIdx.x;
  const int lane = tid & 31;
  const int wave = tid >> 5;         // wave owns rows [wave*32, wave*32+32)
  const int l16  = lane & 15;
  const int half = lane >> 4;

  const unsigned short* qbase = Q  + (size_t)b * T_ * E_ + (size_t)h * DK;
  const unsigned short* kbase = Km + (size_t)b * T_ * E_ + (size_t)h * DK;

#if USE_TDM
  if (tid < 32) {  // one DMA per tile: 128 rows x 128 bf16, global row stride E
    tdm_load_2d_bf16(qbase, lds_addr_of(&q_lds[0][0]), 128, 128, E_, 0, 0, 0);
    tdm_load_2d_bf16(kbase, lds_addr_of(&k_lds[0][0]), 128, 128, E_, 0, 0, 0);
    __builtin_amdgcn_s_wait_tensorcnt(0);
  }
  __syncthreads();
#else
#pragma unroll
  for (int i = 0; i < 16; i += 2) {  // batch pairs: loads first, then stores
    const int ca = i * 128 + tid, cb = (i + 1) * 128 + tid;
    const int ra = ca >> 4, pa = ca & 15;
    const int rb = cb >> 4, pbx = cb & 15;
    U128 qa = *reinterpret_cast<const U128*>(qbase + (size_t)ra * E_ + pa * 8);
    U128 qb2 = *reinterpret_cast<const U128*>(qbase + (size_t)rb * E_ + pbx * 8);
    U128 ka = *reinterpret_cast<const U128*>(kbase + (size_t)ra * E_ + pa * 8);
    U128 kb2 = *reinterpret_cast<const U128*>(kbase + (size_t)rb * E_ + pbx * 8);
    *reinterpret_cast<U128*>(&q_lds[ra][pa * 8]) = qa;
    *reinterpret_cast<U128*>(&q_lds[rb][pbx * 8]) = qb2;
    *reinterpret_cast<U128*>(&k_lds[ra][pa * 8]) = ka;
    *reinterpret_cast<U128*>(&k_lds[rb][pbx * 8]) = kb2;
  }
  __syncthreads();
#endif

  v8f acc[2][8];
#pragma unroll
  for (int i = 0; i < 2; ++i)
#pragma unroll
    for (int j = 0; j < 8; ++j)
      acc[i][j] = v8f{0.f, 0.f, 0.f, 0.f, 0.f, 0.f, 0.f, 0.f};

#pragma unroll
  for (int kk = 0; kk < 4; ++kk) {   // K = DK = 128 in 4 steps of 32
    FragAB af[2], bf[8];
#pragma unroll
    for (int ms = 0; ms < 2; ++ms) {
      const int m = wave * 32 + ms * 16 + l16;
      const unsigned short* p = &q_lds[m][kk * 32 + half * 8];
      af[ms].u[0] = *reinterpret_cast<const U128*>(p);
      af[ms].u[1] = *reinterpret_cast<const U128*>(p + 16);
    }
#pragma unroll
    for (int ns = 0; ns < 8; ++ns) { // scores B = k^T -> k rows already K-contiguous
      const int n = ns * 16 + l16;
      const unsigned short* p = &k_lds[n][kk * 32 + half * 8];
      bf[ns].u[0] = *reinterpret_cast<const U128*>(p);
      bf[ns].u[1] = *reinterpret_cast<const U128*>(p + 16);
    }
#pragma unroll
    for (int ms = 0; ms < 2; ++ms)
#pragma unroll
      for (int ns = 0; ns < 8; ++ns)
        acc[ms][ns] = __builtin_amdgcn_wmma_f32_16x16x32_bf16(
            false, af[ms].v, false, bf[ns].v, (short)0, acc[ms][ns], false, false);
  }

  const float scale = 0.08838834764831845f;  // 1/sqrt(128)
  // Softmax: row (ms,r) lives entirely in one 16-lane half (C layout), one value
  // per lane per n-subtile -> 8 locals + shfl_xor width-16 reductions.
#pragma unroll
  for (int ms = 0; ms < 2; ++ms) {
#pragma unroll
    for (int r = 0; r < 8; ++r) {
      float v[8];
      float mx = -3.4e38f;
#pragma unroll
      for (int j = 0; j < 8; ++j) { v[j] = acc[ms][j][r] * scale; mx = fmaxf(mx, v[j]); }
#pragma unroll
      for (int x = 1; x < 16; x <<= 1) mx = fmaxf(mx, __shfl_xor(mx, x, 16));
      float s = 0.f;
#pragma unroll
      for (int j = 0; j < 8; ++j) { v[j] = __expf(v[j] - mx); s += v[j]; }
#pragma unroll
      for (int x = 1; x < 16; x <<= 1) s += __shfl_xor(s, x, 16);
      const float inv = 1.f / s;
      const int row = wave * 32 + ms * 16 + half * 8 + r;
      unsigned short* prow = P + (size_t)(b * T_ + row) * E_ + (size_t)h * T_;
#pragma unroll
      for (int j = 0; j < 8; ++j)
        prow[j * 16 + l16] = f32_to_bf16_rne(v[j] * inv);
    }
  }
}

// ------------------------------------------------ residual + LayerNorm (fused)
__global__ __launch_bounds__(256)
void ln_residual_kernel(const float* __restrict__ X, const float* __restrict__ Y,
                        const float* __restrict__ G, const float* __restrict__ Bt,
                        float* __restrict__ outF, unsigned short* __restrict__ outB,
                        int E_) {
  __shared__ float rs[256], rq[256];
  const int row = blockIdx.x;
  const int tid = threadIdx.x;
  const float* xr = X + (size_t)row * E_;
  const float* yr = Y + (size_t)row * E_;
  float s = 0.f, q = 0.f;
  for (int e = tid; e < E_; e += 256) {
    const float v = xr[e] + yr[e];
    s += v; q += v * v;
  }
  rs[tid] = s; rq[tid] = q;
  __syncthreads();
  for (int off = 128; off > 0; off >>= 1) {
    if (tid < off) { rs[tid] += rs[tid + off]; rq[tid] += rq[tid + off]; }
    __syncthreads();
  }
  const float invE = 1.f / (float)E_;
  const float mu   = rs[0] * invE;
  const float var  = rq[0] * invE - mu * mu;
  const float rstd = rsqrtf(var + 1e-5f);
  for (int e = tid; e < E_; e += 256) {
    const float v = (xr[e] + yr[e] - mu) * rstd * G[e] + Bt[e];
    if (outF) outF[(size_t)row * E_ + e] = v;
    if (outB) outB[(size_t)row * E_ + e] = f32_to_bf16_rne(v);
  }
}

// ------------------------------------------------------------------- dispatcher
extern "C" void kernel_launch(void* const* d_in, const int* in_sizes, int n_in,
                              void* d_out, int out_size, void* d_ws, size_t ws_size,
                              hipStream_t stream) {
  (void)in_sizes; (void)n_in; (void)out_size; (void)ws_size;
  const int Bb = 32, T_ = 128, E_ = 4096, H_ = 32, F_ = 16384;
  const int Mrows = Bb * T_;  // 4096

  const float* x   = (const float*)d_in[0];
  const float* wq  = (const float*)d_in[1];
  const float* bq  = (const float*)d_in[2];
  const float* wk  = (const float*)d_in[3];
  const float* bk_ = (const float*)d_in[4];
  // d_in[5]=wv, d_in[6]=bv: dead in the reference (attention returns probs only)
  const float* wo  = (const float*)d_in[7];
  const float* bo  = (const float*)d_in[8];
  const float* w1  = (const float*)d_in[9];
  const float* b1  = (const float*)d_in[10];
  const float* w2  = (const float*)d_in[11];
  const float* b2  = (const float*)d_in[12];
  const float* g1  = (const float*)d_in[13];
  const float* be1 = (const float*)d_in[14];
  const float* g2  = (const float*)d_in[15];
  const float* be2 = (const float*)d_in[16];

  char* w = (char*)d_ws;
  size_t off = 0;
  auto alloc = [&](size_t bytes) -> void* {
    void* p = w + off;
    off = (off + bytes + 255) & ~(size_t)255;
    return p;
  };
  unsigned short* xb  = (unsigned short*)alloc((size_t)Mrows * E_ * 2);
  unsigned short* wqT = (unsigned short*)alloc((size_t)E_ * E_ * 2);
  unsigned short* wkT = (unsigned short*)alloc((size_t)E_ * E_ * 2);
  unsigned short* woT = (unsigned short*)alloc((size_t)E_ * E_ * 2);
  unsigned short* w1T = (unsigned short*)alloc((size_t)E_ * F_ * 2);
  unsigned short* w2T = (unsigned short*)alloc((size_t)F_ * E_ * 2);
  unsigned short* qb  = (unsigned short*)alloc((size_t)Mrows * E_ * 2);
  unsigned short* kb  = (unsigned short*)alloc((size_t)Mrows * E_ * 2);
  unsigned short* pb  = (unsigned short*)alloc((size_t)Mrows * E_ * 2);
  float*          atF = (float*)alloc((size_t)Mrows * E_ * 4);
  float*          x1f = (float*)alloc((size_t)Mrows * E_ * 4);
  unsigned short* x1b = (unsigned short*)alloc((size_t)Mrows * E_ * 2);
  unsigned short* h1b = (unsigned short*)alloc((size_t)Mrows * F_ * 2);
  float*          ffn = (float*)alloc((size_t)Mrows * E_ * 4);

  // 1) x -> bf16
  {
    const size_t n4 = (size_t)Mrows * E_ / 4;
    cvt_f32_to_bf16_kernel<<<dim3((unsigned)((n4 + 255) / 256)), 256, 0, stream>>>(x, xb, n4);
  }
  // 2) weights -> transposed bf16 (one pass each; required traffic anyway)
  transpose_cvt_kernel<<<dim3(E_ / 32, E_ / 32), dim3(32, 8), 0, stream>>>(wq, wqT, E_, E_);
  transpose_cvt_kernel<<<dim3(E_ / 32, E_ / 32), dim3(32, 8), 0, stream>>>(wk, wkT, E_, E_);
  transpose_cvt_kernel<<<dim3(E_ / 32, E_ / 32), dim3(32, 8), 0, stream>>>(wo, woT, E_, E_);
  transpose_cvt_kernel<<<dim3(F_ / 32, E_ / 32), dim3(32, 8), 0, stream>>>(w1, w1T, E_, F_);
  transpose_cvt_kernel<<<dim3(E_ / 32, F_ / 32), dim3(32, 8), 0, stream>>>(w2, w2T, F_, E_);
  // 3) q, k projections (bf16 outputs for the attention GEMMs)
  wmma_gemm_bf16_kernel<<<dim3(E_ / 128, Mrows / 128), 256, 0, stream>>>(
      xb, wqT, bq, nullptr, qb, Mrows, E_, E_, 0);
  wmma_gemm_bf16_kernel<<<dim3(E_ / 128, Mrows / 128), 256, 0, stream>>>(
      xb, wkT, bk_, nullptr, kb, Mrows, E_, E_, 0);
  // 4) fused scores + softmax -> probs laid out directly as (B*T, E) for @wo
  attn_scores_softmax_kernel<<<dim3(Bb * H_), 128, 0, stream>>>(qb, kb, pb, T_, E_, H_);
  // 5) attn_out = probs @ wo + bo (f32, feeds LN1)
  wmma_gemm_bf16_kernel<<<dim3(E_ / 128, Mrows / 128), 256, 0, stream>>>(
      pb, woT, bo, atF, nullptr, Mrows, E_, E_, 0);
  // 6) x1 = LN(x + attn_out) -> f32 (residual) + bf16 (next GEMM A)
  ln_residual_kernel<<<dim3(Mrows), 256, 0, stream>>>(x, atF, g1, be1, x1f, x1b, E_);
  // 7) h1 = relu(x1 @ w1 + b1) -> bf16
  wmma_gemm_bf16_kernel<<<dim3(F_ / 128, Mrows / 128), 256, 0, stream>>>(
      x1b, w1T, b1, nullptr, h1b, Mrows, F_, E_, 1);
  // 8) ffn = h1 @ w2 + b2 -> f32
  wmma_gemm_bf16_kernel<<<dim3(E_ / 128, Mrows / 128), 256, 0, stream>>>(
      h1b, w2T, b2, ffn, nullptr, Mrows, E_, F_, 0);
  // 9) out = LN(x1 + ffn)
  ln_residual_kernel<<<dim3(Mrows), 256, 0, stream>>>(x1f, ffn, g2, be2,
                                                      (float*)d_out, nullptr, E_);
}
